// SoftmaxAttention_45878840656618
// MI455X (gfx1250) — matrix operand order repaired and checked by reading, not compiled, not run on hardware
//
#include <hip/hip_runtime.h>
#include <hip/hip_bf16.h>
#include <math.h>

// ---------------------------------------------------------------------------
// MHA forward for MI455X (gfx1250, wave32).  bf16 WMMA, fp32 accumulate.
//   B=4, T=2048, D=2048, H=16, Dh=128
// Pipeline: pack(bf16) -> Q/K/V proj GEMMs (LDS-staged weights) -> V transpose
//           -> flash attention (S^T trick, 32 queries/wave) -> O proj GEMM.
// Workspace (bf16), ~201 MB: xb | wq wk wv wo | Qb | Kb | Vb | Vt ; Ob aliases Vb.
// ---------------------------------------------------------------------------

typedef __bf16 bf16_t;
typedef __attribute__((ext_vector_type(16))) __bf16 v16bf;
typedef __attribute__((ext_vector_type(8)))  __bf16 v8bf;
typedef __attribute__((ext_vector_type(8)))  float   v8f;

#define DEVI static __device__ __forceinline__

static constexpr int BB = 4, TT = 2048, DD = 2048, HH = 16, DH = 128;

DEVI v8f vzero8() {
    v8f z;
#pragma unroll
    for (int i = 0; i < 8; ++i) z[i] = 0.0f;
    return z;
}

DEVI v16bf combine16(v8bf lo, v8bf hi) {
    v16bf r;
#pragma unroll
    for (int i = 0; i < 8; ++i) { r[i] = lo[i]; r[8 + i] = hi[i]; }
    return r;
}

// A-matrix fragment (16x32 bf16, MxK). Lane L: row M = L&15; K chunks
// kb + (L>>4)*8 + {0..7} and kb + 16 + (L>>4)*8 + {0..7}.  Two 16B loads.
DEVI v16bf load_a_frag(const bf16_t* base, int ld, int row0, int kb, int lane) {
    const bf16_t* p = base + (size_t)(row0 + (lane & 15)) * ld + kb + ((lane >> 4) * 8);
    return combine16(*(const v8bf*)p, *(const v8bf*)(p + 16));
}

DEVI v8f wmma_bf16(v16bf a, v16bf b, v8f c) {
    return __builtin_amdgcn_wmma_f32_16x16x32_bf16(
        false, a, false, b, (short)0, c, false, false);
}

DEVI unsigned pack_bf16x2(float lo, float hi) {
    union { bf16_t h[2]; unsigned u; } cv;
    cv.h[0] = (bf16_t)lo; cv.h[1] = (bf16_t)hi;
    return cv.u;
}

// ---------------------------------------------------------------------------
// fp32 -> bf16 pack (grid-stride)
// ---------------------------------------------------------------------------
__global__ void __launch_bounds__(256)
pack_bf16(const float* __restrict__ src, bf16_t* __restrict__ dst, long long n) {
    long long i = (long long)blockIdx.x * blockDim.x + threadIdx.x;
    long long stride = (long long)gridDim.x * blockDim.x;
    for (; i < n; i += stride) dst[i] = (bf16_t)src[i];
}

// ---------------------------------------------------------------------------
// C[M,N] = A[M,K] @ W[N,K]^T (bf16 in, fp32 accum).  Block tile 256x64:
// 8 waves (32 rows each) share a 64-col weight strip double-buffered in LDS.
// ---------------------------------------------------------------------------
static constexpr int WST = 40;  // padded LDS row stride (bf16 elems; 80B -> conflict-free)

template <bool OUT_F32>
__global__ void __launch_bounds__(256)
gemm_nt(const bf16_t* __restrict__ A, const bf16_t* __restrict__ W,
        void* __restrict__ Cout, int M, int N, int K) {
    __shared__ __align__(32) bf16_t wt[2][64 * WST];

    const int lane = threadIdx.x & 31;
    const int wsl  = threadIdx.x >> 5;
    const int ntn  = N / 64;
    const int n0   = (blockIdx.x % ntn) * 64;
    const int m0   = (blockIdx.x / ntn) * 256 + wsl * 32;

    // cooperative weight-tile stage: 64 rows x 32 k, 16B per thread, coalesced
    const int srow = threadIdx.x >> 2, sch = (threadIdx.x & 3) * 8;

    *(v8bf*)&wt[0][srow * WST + sch] = *(const v8bf*)(W + (size_t)(n0 + srow) * K + sch);
    __syncthreads();

    v8f acc[2][4];
#pragma unroll
    for (int i = 0; i < 2; ++i)
#pragma unroll
        for (int j = 0; j < 4; ++j) acc[i][j] = vzero8();

    int ib = 0;
    for (int kb = 0; kb < K; kb += 32, ib ^= 1) {
        if (kb + 32 < K)
            *(v8bf*)&wt[ib ^ 1][srow * WST + sch] =
                *(const v8bf*)(W + (size_t)(n0 + srow) * K + kb + 32 + sch);

        v16bf a0 = load_a_frag(A, K, m0,      kb, lane);
        v16bf a1 = load_a_frag(A, K, m0 + 16, kb, lane);
        v16bf bfrag[4];
#pragma unroll
        for (int j = 0; j < 4; ++j) {
            const bf16_t* p = &wt[ib][(j * 16 + (lane & 15)) * WST + ((lane >> 4) * 8)];
            bfrag[j] = combine16(*(const v8bf*)p, *(const v8bf*)(p + 16));
        }
#pragma unroll
        for (int j = 0; j < 4; ++j) {
            acc[0][j] = wmma_bf16(a0, bfrag[j], acc[0][j]);
            acc[1][j] = wmma_bf16(a1, bfrag[j], acc[1][j]);
        }
        __syncthreads();
    }

    const int nlo = lane & 15, rhi = (lane >> 4) * 8;
#pragma unroll
    for (int mi = 0; mi < 2; ++mi)
#pragma unroll
        for (int j = 0; j < 4; ++j)
#pragma unroll
            for (int r = 0; r < 8; ++r) {
                size_t idx = (size_t)(m0 + mi * 16 + rhi + r) * N + (n0 + j * 16 + nlo);
                if (OUT_F32) ((float*)Cout)[idx] = acc[mi][j][r];
                else         ((bf16_t*)Cout)[idx] = (bf16_t)acc[mi][j][r];
            }
}

// ---------------------------------------------------------------------------
// Vt[b][d][t] = V[b*T + t][d]   (32x32 LDS tile transpose, bf16)
// ---------------------------------------------------------------------------
__global__ void __launch_bounds__(256)
transpose_v(const bf16_t* __restrict__ V, bf16_t* __restrict__ Vt) {
    __shared__ bf16_t tile[32][33];
    const int b = blockIdx.z;
    const int t0 = blockIdx.x * 32, d0 = blockIdx.y * 32;
    const int x = threadIdx.x, y = threadIdx.y;   // blockDim = (32, 8)
#pragma unroll
    for (int i = 0; i < 32; i += 8)
        tile[y + i][x] = V[(size_t)(b * TT + t0 + y + i) * DD + d0 + x];
    __syncthreads();
#pragma unroll
    for (int i = 0; i < 32; i += 8)
        Vt[(size_t)b * DD * TT + (size_t)(d0 + y + i) * TT + t0 + x] = tile[x][y + i];
}

// ---------------------------------------------------------------------------
// Flash attention, transposed-score formulation.  One wave per (b,h,32-query
// tile = two 16-query sets sharing all K/V fragments), 32-key blocks, causal.
//   S^T = K (A-frag) @ Q^T (B-frag): queries on the LANE axis, keys on the
//   VGPR axis -> softmax max/sum are per-lane scalars + ONE xor-16 shuffle.
//   P^T -> B-layout: bf16 pair-pack only, lanes keep their own keys (no LDS).
//   O^T += V^T (A-frag, from Vt) @ P^T; each V frag feeds both query sets.
// ---------------------------------------------------------------------------
__global__ void __launch_bounds__(256)
flash_attn(const bf16_t* __restrict__ Q, const bf16_t* __restrict__ Kb,
           const bf16_t* __restrict__ Vt, bf16_t* __restrict__ O) {
    const int lane = threadIdx.x & 31;
    const int hb   = lane >> 4;          // key/dh half selector
    const int ql   = lane & 15;          // this lane's query within a 16-tile
    const int wid  = blockIdx.x * 8 + (threadIdx.x >> 5);
    const int qt   = wid & 63;           // T/32 = 64 query tiles
    const int h    = (wid >> 6) & 15;
    const int b    = wid >> 10;
    const int q0   = qt * 32;

    const bf16_t* kbase = Kb + (size_t)(b * TT) * DD + h * DH;                // [t][Dh]
    const bf16_t* vtb   = Vt + (size_t)b * DD * TT + (size_t)(h * DH) * TT;   // [Dh][t]

    int qrow[2] = { q0 + ql, q0 + 16 + ql };

    // Resident Q^T B-fragments, one set per 16-query half-tile
    v16bf qf[2][4];
#pragma unroll
    for (int s = 0; s < 2; ++s) {
        const bf16_t* qptr = Q + (size_t)(b * TT + qrow[s]) * DD + h * DH + hb * 8;
#pragma unroll
        for (int kk = 0; kk < 4; ++kk) {
            const bf16_t* p = qptr + kk * 32;
            qf[s][kk] = combine16(*(const v8bf*)p, *(const v8bf*)(p + 16));
        }
    }

    v8f ot[2][8];                        // O^T accum: d on VGPR axis, query on lane
#pragma unroll
    for (int s = 0; s < 2; ++s)
#pragma unroll
        for (int j = 0; j < 8; ++j) ot[s][j] = vzero8();
    float m[2] = { -INFINITY, -INFINITY }, l[2] = { 0.0f, 0.0f };

    const float scale = 0.08838834764831845f;  // 1/sqrt(128)
    const int nblocks = q0 / 32 + 1;           // both query sets need the same range

    for (int blk = 0; blk < nblocks; ++blk) {
        const int k0 = blk * 32;

        // S^T[32 keys, 2x16 queries]: K frags shared by both query sets
        v8f st[2][2];
#pragma unroll
        for (int s = 0; s < 2; ++s) { st[s][0] = vzero8(); st[s][1] = vzero8(); }
#pragma unroll
        for (int kk = 0; kk < 4; ++kk) {
            v16bf ka0 = load_a_frag(kbase, DD, k0,      kk * 32, lane);
            v16bf ka1 = load_a_frag(kbase, DD, k0 + 16, kk * 32, lane);
#pragma unroll
            for (int s = 0; s < 2; ++s) {
                st[s][0] = wmma_bf16(ka0, qf[s][kk], st[s][0]);
                st[s][1] = wmma_bf16(ka1, qf[s][kk], st[s][1]);
            }
        }

        union { v16bf v; unsigned u[8]; } pfrag[2];

#pragma unroll
        for (int s = 0; s < 2; ++s) {
#pragma unroll
            for (int r = 0; r < 8; ++r) { st[s][0][r] *= scale; st[s][1][r] *= scale; }

            // Causal mask (only last block crosses the diagonal).
            // st[s][0] keys: k0 + r + 8*hb ; st[s][1]: +16 ; query = qrow[s].
            if (blk == nblocks - 1) {
                const int kb0 = k0 + 8 * hb;
#pragma unroll
                for (int r = 0; r < 8; ++r) {
                    if (kb0 + r      > qrow[s]) st[s][0][r] = -3.0e38f;
                    if (kb0 + 16 + r > qrow[s]) st[s][1][r] = -3.0e38f;
                }
            }

            // Per-lane max over 16 local keys + one lane-pair shuffle
            float v = fmaxf(st[s][0][0], st[s][1][0]);
#pragma unroll
            for (int r = 1; r < 8; ++r) v = fmaxf(v, fmaxf(st[s][0][r], st[s][1][r]));
            v = fmaxf(v, __shfl_xor(v, 16, 32));
            const float mnew  = fmaxf(m[s], v);
            const float alpha = __expf(m[s] - mnew);
            m[s] = mnew;
            l[s] *= alpha;
#pragma unroll
            for (int j = 0; j < 8; ++j)
#pragma unroll
                for (int r = 0; r < 8; ++r) ot[s][j][r] *= alpha;

            // P = exp(S - m); per-lane partial sum + one shuffle
            float sum = 0.0f;
#pragma unroll
            for (int r = 0; r < 8; ++r) {
                st[s][0][r] = __expf(st[s][0][r] - m[s]);
                st[s][1][r] = __expf(st[s][1][r] - m[s]);
                sum += st[s][0][r] + st[s][1][r];
            }
            sum += __shfl_xor(sum, 16, 32);
            l[s] += sum;

            // Build P^T B-frag (32x16): pack key pairs, kept in own lane (sw0/sw1 unused)
            unsigned pk0[4], pk1[4], sw0[4], sw1[4];
#pragma unroll
            for (int q = 0; q < 4; ++q) {
                pk0[q] = pack_bf16x2(st[s][0][2 * q], st[s][0][2 * q + 1]);
                pk1[q] = pack_bf16x2(st[s][1][2 * q], st[s][1][2 * q + 1]);
            }
#pragma unroll
            for (int q = 0; q < 4; ++q) {
                sw0[q] = (unsigned)__shfl_xor((int)pk0[q], 16, 32);
                sw1[q] = (unsigned)__shfl_xor((int)pk1[q], 16, 32);
            }
#pragma unroll
            for (int q = 0; q < 4; ++q) {
                pfrag[s].u[q]     = pk0[q];   // keys hb*8 + 0..7
                pfrag[s].u[4 + q] = pk1[q];   // keys 16 + hb*8 + 0..7
            }
        }

        // O^T += V^T @ P^T : each Vt A-frag feeds both query sets
#pragma unroll
        for (int j = 0; j < 8; ++j) {
            v16bf va = load_a_frag(vtb, TT, j * 16, k0, lane);
#pragma unroll
            for (int s = 0; s < 2; ++s)
                ot[s][j] = wmma_bf16(va, pfrag[s].v, ot[s][j]);
        }
    }

    // Normalize and store: per lane one packed 16B store per d-tile per set
#pragma unroll
    for (int s = 0; s < 2; ++s) {
        const float inv = 1.0f / l[s];
        bf16_t* orow = O + (size_t)(b * TT + qrow[s]) * DD + h * DH + hb * 8;
#pragma unroll
        for (int j = 0; j < 8; ++j) {
            v8bf ob;
#pragma unroll
            for (int r = 0; r < 8; ++r) ob[r] = (bf16_t)(ot[s][j][r] * inv);
            *(v8bf*)(orow + j * 16) = ob;
        }
    }
}

// ---------------------------------------------------------------------------
// Host-side launch
// ---------------------------------------------------------------------------
extern "C" void kernel_launch(void* const* d_in, const int* in_sizes, int n_in,
                              void* d_out, int out_size, void* d_ws, size_t ws_size,
                              hipStream_t stream) {
    (void)in_sizes; (void)n_in; (void)out_size; (void)ws_size;
    const float* x  = (const float*)d_in[0];
    const float* Wq = (const float*)d_in[1];
    const float* Wk = (const float*)d_in[2];
    const float* Wv = (const float*)d_in[3];
    const float* Wo = (const float*)d_in[4];

    const long long nx = (long long)BB * TT * DD;   // 16,777,216
    const long long nw = (long long)DD * DD;        //  4,194,304

    bf16_t* p   = (bf16_t*)d_ws;
    bf16_t* xb  = p; p += nx;
    bf16_t* wqb = p; p += nw;
    bf16_t* wkb = p; p += nw;
    bf16_t* wvb = p; p += nw;
    bf16_t* wob = p; p += nw;
    bf16_t* Qb  = p; p += nx;
    bf16_t* Kbf = p; p += nx;
    bf16_t* Vb  = p; p += nx;
    bf16_t* Vt  = p; p += nx;
    bf16_t* Ob  = Vb;                // alias: V dead after transpose

    // 1) fp32 -> bf16
    pack_bf16<<<2048, 256, 0, stream>>>(x,  xb,  nx);
    pack_bf16<<<1024, 256, 0, stream>>>(Wq, wqb, nw);
    pack_bf16<<<1024, 256, 0, stream>>>(Wk, wkb, nw);
    pack_bf16<<<1024, 256, 0, stream>>>(Wv, wvb, nw);
    pack_bf16<<<1024, 256, 0, stream>>>(Wo, wob, nw);

    // 2) Q/K/V projections: [8192,2048] = [8192,2048] @ [2048,2048]^T
    const int M = BB * TT, N = DD, K = DD;
    const int gemm_blocks = (M / 256) * (N / 64);   // 256x64 block tiles
    gemm_nt<false><<<gemm_blocks, 256, 0, stream>>>(xb, wqb, Qb,  M, N, K);
    gemm_nt<false><<<gemm_blocks, 256, 0, stream>>>(xb, wkb, Kbf, M, N, K);
    gemm_nt<false><<<gemm_blocks, 256, 0, stream>>>(xb, wvb, Vb,  M, N, K);

    // 3) V -> V^T (per batch: [D][T])
    dim3 tg(TT / 32, DD / 32, BB);
    transpose_v<<<tg, dim3(32, 8), 0, stream>>>(Vb, Vt);

    // 4) Flash attention: B*H*(T/32) = 4096 waves, 8 per block
    flash_attn<<<(BB * HH * (TT / 32)) / 8, 256, 0, stream>>>(Qb, Kbf, Vt, Ob);

    // 5) Output projection, fp32 straight to d_out
    gemm_nt<true><<<gemm_blocks, 256, 0, stream>>>(Ob, wob, (float*)d_out, M, N, K);
}
